// minerva_transform_60000693125637
// MI455X (gfx1250) — matrix operand order repaired and checked by reading, not compiled, not run on hardware
//
#include <hip/hip_runtime.h>
#include <math.h>

// ---------------------------------------------------------------------------
// Minerva transform, fused for MI455X (gfx1250, wave32, WMMA bf16 + TDM).
//   logits[b] = sum_n ( (Xn[b]·Dn[n])^3 * rh[n] ),  preds = sigmoid(logits)
// Xn/Dn are L2-normalized embeddings kept in bf16 in workspace (L2-resident).
// Big GEMM: B panel (64 cols x 512 K, bf16) staged once per block into LDS by
// the Tensor Data Mover (padded rows for bank-conflict-free ds_load_b128).
// ---------------------------------------------------------------------------

typedef __attribute__((ext_vector_type(16))) __bf16 v16bf;
typedef __attribute__((ext_vector_type(8)))  __bf16 v8bf;
typedef __attribute__((ext_vector_type(8)))  float  v8f;
typedef __attribute__((ext_vector_type(4)))  unsigned int u32x4;
typedef __attribute__((ext_vector_type(8)))  int          i32x8;
typedef __attribute__((ext_vector_type(4)))  int          i32x4;

#define B_ROWS 4096
#define N_ROWS 20000
#define N_PAD  20032   // 313 * 64
#define F_DIM  768
#define E_DIM  512

// LDS B panel: 64 rows x 512 bf16, padded +8 elements (16 B) per row.
// Row stride = 1040 B = 260 DWORDs -> 4-bank shift per row: lanes 0..15
// reading rows r at the same k hit banks {4r..4r+3} mod 64 -> conflict-free.
#define LDSB_STRIDE 520
#define LDSB_BYTES  (64 * LDSB_STRIDE * 2)   // 66,560 B (dynamic LDS)

#if defined(__HIP_DEVICE_COMPILE__) && defined(__has_builtin)
#if __has_builtin(__builtin_amdgcn_tensor_load_to_lds) && \
    __has_builtin(__builtin_amdgcn_s_wait_tensorcnt)
#define MINERVA_USE_TDM 1
#endif
#endif
#ifndef MINERVA_USE_TDM
#define MINERVA_USE_TDM 0
#endif

union FragBF { v16bf v; v8bf h[2]; };

// ----- bf16 fragment loads (row-major source, ld = row stride in elements) --
// A 16x32 bf16 layout: lane<16 holds row M=lane, K = k0..k0+15 contiguous;
// lane>=16 holds same rows with K = k0+16..k0+31.
__device__ __forceinline__ v16bf load_a_bf16(const __bf16* base, int ld,
                                             int row0, int k0, int lane) {
  const int half = lane >> 4, lr = lane & 15;
  const __bf16* p = base + (size_t)(row0 + lr) * ld + k0 + half * 16;
  FragBF f;
  f.h[0] = *(const v8bf*)(p);        // K = k0 + half*16 + [0..7]
  f.h[1] = *(const v8bf*)(p + 8);    // K = k0 + half*16 + 8 + [0..7]
  return f.v;
}
// B 32x16 bf16: lane<16 holds column N=lane with K = k0..k0+15 contiguous,
// lane>=16 holds K = k0+16..k0+31 (half-wave contiguous-K convention).
__device__ __forceinline__ v16bf load_b_lds(const __bf16* sB, int col0,
                                            int k0, int lane) {
  const int half = lane >> 4, lr = lane & 15;
  const __bf16* p = sB + (col0 + lr) * LDSB_STRIDE + k0 + half * 16;
  FragBF f;
  f.h[0] = *(const v8bf*)(p);        // ds_load_b128
  f.h[1] = *(const v8bf*)(p + 8);
  return f.v;
}

// ----- f32 -> bf16 fragment builders (for the embedding GEMMs) -------------
__device__ __forceinline__ v16bf cvt16_to_frag(const float* a, const float* b) {
  float4 x0 = ((const float4*)a)[0];
  float4 x1 = ((const float4*)a)[1];
  float4 y0 = ((const float4*)b)[0];
  float4 y1 = ((const float4*)b)[1];
  float t[16] = {x0.x, x0.y, x0.z, x0.w, x1.x, x1.y, x1.z, x1.w,
                 y0.x, y0.y, y0.z, y0.w, y1.x, y1.y, y1.z, y1.w};
  FragBF f;
#pragma unroll
  for (int j = 0; j < 16; ++j) f.v[j] = (__bf16)t[j];
  return f.v;
}
__device__ __forceinline__ v16bf load_a_f32(const float* base, int ld,
                                            int row0, int k0, int lane) {
  const int half = lane >> 4, lr = lane & 15;
  const float* p = base + (size_t)(row0 + lr) * ld + k0 + half * 16;
  return cvt16_to_frag(p, p + 8);
}
__device__ __forceinline__ v16bf load_b_f32(const float* baseT, int ld,
                                            int col0, int k0, int lane) {
  const int half = lane >> 4, lr = lane & 15;
  const float* p = baseT + (size_t)(col0 + lr) * ld + k0 + half * 16;
  return cvt16_to_frag(p, p + 8);
}

__device__ __forceinline__ float half16_reduce(float v) {
  v += __shfl_xor(v, 1, 32);
  v += __shfl_xor(v, 2, 32);
  v += __shfl_xor(v, 4, 32);
  v += __shfl_xor(v, 8, 32);   // stays within each 16-lane half of the wave32
  return v;
}

// ---------------------------------------------------------------------------
// Kernel 0: zero logits accumulator, zero Dn pad rows [20000,20032), rh pad.
// ---------------------------------------------------------------------------
__global__ __launch_bounds__(256) void init_ws_kernel(float* __restrict__ logits,
                                                      unsigned int* __restrict__ dn_pad,
                                                      float* __restrict__ rh_pad) {
  int i = blockIdx.x * blockDim.x + threadIdx.x;
  if (i < B_ROWS) logits[i] = 0.0f;
  if (i < (N_PAD - N_ROWS) * E_DIM / 2) dn_pad[i] = 0u;  // 32*512 bf16 = 8192 u32
  if (i < (N_PAD - N_ROWS)) rh_pad[i] = 0.0f;
}

// ---------------------------------------------------------------------------
// Kernel 1/2: embed + bias + row-L2-normalize, store bf16.
//   dst[m] = normalize(src[m] @ gw^T + gb)
// One block per 16-row strip; wave w covers output cols [w*64, w*64+64).
// ---------------------------------------------------------------------------
__global__ __launch_bounds__(256) void embed_norm_kernel(
    const float* __restrict__ src, const float* __restrict__ gw,
    const float* __restrict__ gb, __bf16* __restrict__ dst,
    const float* __restrict__ r, const float* __restrict__ hw,
    const float* __restrict__ hb, float* __restrict__ rh, int do_rh) {
  __shared__ float s_sumsq[16];
  const int tid  = threadIdx.x;
  const int wave = tid >> 5, lane = tid & 31;
  const int half = lane >> 4, lr = lane & 15;
  const int m0   = blockIdx.x * 16;
  const int nc0  = wave * 64;

  if (do_rh && tid < 16) {  // rh = (2r-1)*h_w + h_b  (C == 1)
    int row = m0 + tid;
    rh[row] = (r[row] * 2.0f - 1.0f) * hw[0] + hb[0];
  }
  if (tid < 16) s_sumsq[tid] = 0.0f;
  __syncthreads();

  v8f acc[4] = {};
  for (int k0 = 0; k0 < F_DIM; k0 += 32) {
    v16bf a = load_a_f32(src, F_DIM, m0, k0, lane);
#pragma unroll
    for (int t = 0; t < 4; ++t) {
      v16bf b = load_b_f32(gw, F_DIM, nc0 + t * 16, k0, lane);
      acc[t] = __builtin_amdgcn_wmma_f32_16x16x32_bf16(
          false, a, false, b, (short)0, acc[t], false, false);
    }
  }

  float vals[4][8];
  float sq[8];
#pragma unroll
  for (int i = 0; i < 8; ++i) sq[i] = 0.0f;
#pragma unroll
  for (int t = 0; t < 4; ++t) {
    float bias = gb[nc0 + t * 16 + lr];
#pragma unroll
    for (int i = 0; i < 8; ++i) {
      float v = acc[t][i] + bias;   // row m0+half*8+i, col nc0+t*16+lr
      vals[t][i] = v;
      sq[i] += v * v;
    }
  }
#pragma unroll
  for (int i = 0; i < 8; ++i) {
    float s = half16_reduce(sq[i]);
    if (lr == 0) atomicAdd(&s_sumsq[half * 8 + i], s);  // ds_add_f32
  }
  __syncthreads();

#pragma unroll
  for (int i = 0; i < 8; ++i) {
    int row = m0 + half * 8 + i;
    float inv = 1.0f / fmaxf(sqrtf(s_sumsq[half * 8 + i]), 1e-12f);
#pragma unroll
    for (int t = 0; t < 4; ++t)
      dst[(size_t)row * E_DIM + nc0 + t * 16 + lr] = (__bf16)(vals[t][i] * inv);
  }
}

// ---------------------------------------------------------------------------
// Kernel 3: fused cosine-sim GEMM + cube + rh-weighted reduction over N.
// Block: 8 waves x (32 rows each) = 256 rows x 64 cols. The 64x512 bf16 Dn
// panel (shared by all 8 waves) is staged in LDS once per block via the
// Tensor Data Mover (TENSORcnt), with D# padding for conflict-free reads.
// Each wave: 8 WMMAs per 32-K step (2 M-tiles x 4 N-tiles); epilogue cubes,
// weights by rh[n] and reduces -> 32 global_atomic_add_f32 per wave.
// ---------------------------------------------------------------------------
__global__ __launch_bounds__(256) void minerva_gemm_kernel(
    const __bf16* __restrict__ Xn, const __bf16* __restrict__ Dn,
    const float* __restrict__ rh, float* __restrict__ logits) {
  extern __shared__ __bf16 sB[];                 // 64 x 520 bf16 (66,560 B)
  const int tid  = threadIdx.x;
  const int wave = tid >> 5, lane = tid & 31;
  const int half = lane >> 4, lr = lane & 15;
  const int m0 = (blockIdx.y * 8 + wave) * 32;   // 16 * 8 * 32 = 4096 rows
  const int n0 = blockIdx.x * 64;                // 313 groups = 20032 padded cols

#if MINERVA_USE_TDM
  if (wave == 0) {
    // D#: 2D tile 512 x 64 of 2-byte elements, source row stride 512,
    // LDS pad: after every 256 DWORDs (1024 B row) insert 4 DWORDs (16 B).
    unsigned lds_addr = (unsigned)(size_t)(void*)sB;
    unsigned long long ga =
        (unsigned long long)(size_t)(Dn + (size_t)n0 * E_DIM);
    u32x4 g0;
    g0[0] = 1u;                                   // count=1, user mode
    g0[1] = lds_addr;                             // LDS dest (bytes)
    g0[2] = (unsigned)(ga & 0xFFFFFFFFu);         // global_addr[31:0]
    g0[3] = (unsigned)((ga >> 32) & 0x01FFFFFFu)  // global_addr[56:32]
            | (2u << 30);                         // type=2 ("image")
    i32x8 g1;
    g1[0] = (1 << 16)        // data_size = 1 -> 2-byte elements
          | (1 << 20)        // pad_enable
          | (7 << 22)        // pad_interval = 7 -> every 256 DWORDs
          | (3 << 25);       // pad_amount  = 3 -> 4 DWORDs (16 B)
    g1[1] = (E_DIM & 0xFFFF) << 16;   // tensor_dim0[15:0] = 512
    g1[2] = (64 << 16);               // tensor_dim0[31:16]=0 | tensor_dim1=64
    g1[3] = (E_DIM << 16);            // tensor_dim1[31:16]=0 | tile_dim0=512
    g1[4] = 64;                       // tile_dim1=64, tile_dim2=0
    g1[5] = E_DIM;                    // tensor_dim0_stride[31:0] = 512
    g1[6] = 0;                        // stride0[47:32]=0 | dim1_stride lo=0
    g1[7] = 0;
    i32x4 gz4 = {0, 0, 0, 0};         // groups 2/3 unused (<=2D tensor)
    i32x8 gz8 = {0, 0, 0, 0, 0, 0, 0, 0};
    // 6-arg toolchain form: (g0, g1, g2, g3, g_extra, cpol)
    __builtin_amdgcn_tensor_load_to_lds(g0, g1, gz4, gz4, gz8, 0);
    __builtin_amdgcn_s_wait_tensorcnt(0);
  }
#else
  // Fallback: cooperative global->LDS copy into the same padded layout.
  for (int i = tid; i < 64 * (E_DIM / 8); i += 256) {   // 16B chunks
    int row = i >> 6, seg = i & 63;
    const uint4 v = ((const uint4*)(Dn + (size_t)(n0 + row) * E_DIM))[seg];
    ((uint4*)(sB + row * LDSB_STRIDE))[seg] = v;
  }
#endif
  __syncthreads();

  v8f acc[2][4] = {};
  for (int k0 = 0; k0 < E_DIM; k0 += 32) {
    v16bf a0 = load_a_bf16(Xn, E_DIM, m0, k0, lane);
    v16bf a1 = load_a_bf16(Xn, E_DIM, m0 + 16, k0, lane);
#pragma unroll
    for (int t = 0; t < 4; ++t) {
      v16bf b = load_b_lds(sB, t * 16, k0, lane);
      acc[0][t] = __builtin_amdgcn_wmma_f32_16x16x32_bf16(
          false, a0, false, b, (short)0, acc[0][t], false, false);
      acc[1][t] = __builtin_amdgcn_wmma_f32_16x16x32_bf16(
          false, a1, false, b, (short)0, acc[1][t], false, false);
    }
  }

  // sign(s)|s|^3 == s^3 (odd power); weight by rh, reduce over this tile's N
#pragma unroll
  for (int u = 0; u < 2; ++u) {
    float part[8];
#pragma unroll
    for (int i = 0; i < 8; ++i) part[i] = 0.0f;
#pragma unroll
    for (int t = 0; t < 4; ++t) {
      float w = rh[n0 + t * 16 + lr];
#pragma unroll
      for (int i = 0; i < 8; ++i) {
        float s = acc[u][t][i];
        part[i] += w * s * s * s;
      }
    }
#pragma unroll
    for (int i = 0; i < 8; ++i) {
      float s = half16_reduce(part[i]);
      if (lr == 0)
        atomicAdd(&logits[m0 + u * 16 + half * 8 + i], s);  // global_atomic_add_f32
    }
  }
}

// ---------------------------------------------------------------------------
// Kernel 4: out[0:4096] = logits, out[4096:8192] = sigmoid(logits)
// ---------------------------------------------------------------------------
__global__ __launch_bounds__(256) void finalize_kernel(const float* __restrict__ logits,
                                                       float* __restrict__ out) {
  int i = blockIdx.x * blockDim.x + threadIdx.x;
  if (i < B_ROWS) {
    float x = logits[i];
    out[i] = x;
    out[B_ROWS + i] = 1.0f / (1.0f + __expf(-x));
  }
}

// ---------------------------------------------------------------------------
extern "C" void kernel_launch(void* const* d_in, const int* in_sizes, int n_in,
                              void* d_out, int out_size, void* d_ws, size_t ws_size,
                              hipStream_t stream) {
  (void)in_sizes; (void)n_in; (void)out_size; (void)ws_size;
  const float* X  = (const float*)d_in[0];  // [4096, 768]
  const float* D  = (const float*)d_in[1];  // [20000, 768]
  const float* r  = (const float*)d_in[2];  // [20000, 1]
  const float* gw = (const float*)d_in[3];  // [512, 768]
  const float* gb = (const float*)d_in[4];  // [512]
  const float* hw = (const float*)d_in[5];  // [1, 1]
  const float* hb = (const float*)d_in[6];  // [1]

  // workspace layout (~24.8 MB, all offsets 256B aligned)
  char* ws = (char*)d_ws;
  __bf16* Xn     = (__bf16*)(ws);                               //  4096*512 bf16
  __bf16* Dn     = (__bf16*)(ws + 4194304);                     // 20032*512 bf16
  float*  rh     = (float*)(ws + 4194304 + 20512768);           // 20032 f32
  float*  logits = (float*)(ws + 4194304 + 20512768 + 80128);   //  4096 f32

  init_ws_kernel<<<32, 256, 0, stream>>>(
      logits, (unsigned int*)(Dn + (size_t)N_ROWS * E_DIM), rh + N_ROWS);

  embed_norm_kernel<<<B_ROWS / 16, 256, 0, stream>>>(
      X, gw, gb, Xn, nullptr, nullptr, nullptr, nullptr, 0);
  embed_norm_kernel<<<N_ROWS / 16, 256, 0, stream>>>(
      D, gw, gb, Dn, r, hw, hb, rh, 1);

  dim3 grid(N_PAD / 64, B_ROWS / 256);   // (313, 16), 8 waves/block
  minerva_gemm_kernel<<<grid, 256, LDSB_BYTES, stream>>>(Xn, Dn, rh, logits);

  finalize_kernel<<<B_ROWS / 256, 256, 0, stream>>>(logits, (float*)d_out);
}
